// LatentODE2_52913997087352
// MI455X (gfx1250) — matrix operand order repaired and checked by reading, not compiled, not run on hardware
//
#include <hip/hip_runtime.h>

typedef __attribute__((ext_vector_type(16))) _Float16 v16h;
typedef __attribute__((ext_vector_type(8)))  _Float16 v8h;
typedef __attribute__((ext_vector_type(8)))  float    v8f;
typedef __attribute__((ext_vector_type(4)))  float    v4f;

// Problem constants (match reference setup_inputs)
constexpr int   BATCH      = 8192;
constexpr int   T          = 100;
constexpr int   D          = 16;
constexpr int   H          = 32;
constexpr int   HD         = 50;       // min((H+D)*2, 50)
constexpr int   KC         = D + H;    // 48 (concat width)
constexpr float DT_SCALER  = 1.0f / 24.0f;
constexpr int   EULER      = 10;
constexpr float STEP       = 0.1f;

constexpr int   PITCH      = 72;       // LDS row pitch in halves (144B: 16B-aligned, bank-staggered)
constexpr int   WPB        = 4;        // waves per block
constexpr int   TILE_H     = 16 * PITCH;        // halves per 16x64 activation tile
constexpr int   WAVE_H     = 3 * TILE_H;        // XH + Z3 + Z1

#define LDS_FENCE() asm volatile("s_wait_dscnt 0" ::: "memory")

// Native CDNA5 transcendental tanh (single TRANS op, co-executes with WMMA pipe)
__device__ __forceinline__ float vtanh(float x) {
    float r;
    asm("v_tanh_f32 %0, %1" : "=v"(r) : "v"(x));
    return r;
}

// B-fragment (K=32 x N=16, f16) for W (Nout x Kin row-major), tile (kt, nt).
// Layout: lanes 0-15 hold column N=lane, VGPR v holds K=2v,2v+1; lanes 16-31 hold upper K half.
__device__ __forceinline__ v16h load_b_frag(const float* __restrict__ W,
                                            int Nout, int Kin, int nt, int kt, int lane) {
    const int n  = nt * 16 + (lane & 15);
    const int kb = kt * 32 + (lane >> 4) * 16;
    v16h b;
#pragma unroll
    for (int v = 0; v < 8; ++v) {
        int k0 = kb + 2 * v;
        float f0 = (n < Nout && (k0     < Kin)) ? W[n * Kin + k0]     : 0.0f;
        float f1 = (n < Nout && (k0 + 1 < Kin)) ? W[n * Kin + k0 + 1] : 0.0f;
        b[2 * v]     = (_Float16)f0;
        b[2 * v + 1] = (_Float16)f1;
    }
    return b;
}

// A-fragment (M=16 x K=32, f16) straight from a packed-f16 LDS tile:
// two 16B ds_load_b128, zero conversions. p = &tile[row*PITCH + kw + hsel*16].
__device__ __forceinline__ v16h load_a_frag(const _Float16* p) {
    v8h lo = *(const v8h*)p;
    v8h hi = *(const v8h*)(p + 8);
    return __builtin_shufflevector(lo, hi, 0, 1, 2, 3, 4, 5, 6, 7,
                                           8, 9, 10, 11, 12, 13, 14, 15);
}

__device__ __forceinline__ v8f wmma16(v16h a, v16h b, v8f c) {
    return __builtin_amdgcn_wmma_f32_16x16x32_f16(false, a, false, b, (short)0, c,
                                                  false, false);
}

__global__ __launch_bounds__(WPB * 32, 1)
void latent_ode_kernel(const float* __restrict__ dt, const float* __restrict__ x,
                       const float* __restrict__ W1, const float* __restrict__ b1,
                       const float* __restrict__ W2, const float* __restrict__ b2,
                       const float* __restrict__ W3, const float* __restrict__ b3,
                       const float* __restrict__ W4, const float* __restrict__ b4,
                       float* __restrict__ out) {
    __shared__ __align__(16) _Float16 smemh[WPB * WAVE_H];
    __shared__ float smemf[WPB * 32];

    const int lane = threadIdx.x & 31;
    const int wv   = threadIdx.x >> 5;
    const int base = (blockIdx.x * WPB + wv) * 16;   // first batch row of this wave's tile
    const int nl   = lane & 15;                      // N column / A row owned by this lane
    const int hsel = lane >> 4;                      // 0/1 K-half select
    const int mb   = hsel * 8;                       // C-fragment row base
    const int row  = nl;                             // tile row for loads

    _Float16* XH = smemh + wv * WAVE_H;   // [16][PITCH] cols 0-15: x, 16-47: h, 48-63: zero
    _Float16* Z3 = XH + TILE_H;           // [16][PITCH] tanh(z3), 64 cols
    _Float16* Z1 = Z3 + TILE_H;           // [16][PITCH] tanh(z1), cols 0..49
    float*    Y  = smemf + wv * 32;       // [16] y carry
    float*    SC = Y + 16;                // [16] per-row dt scale

    // ---- Register-resident weight fragments (loaded once) ----
    v16h B1f[2][4], B2f[2][2], B3f[4], B4f[2];
#pragma unroll
    for (int kt = 0; kt < 2; ++kt)
#pragma unroll
        for (int nt = 0; nt < 4; ++nt) B1f[kt][nt] = load_b_frag(W1, HD, KC, nt, kt, lane);
#pragma unroll
    for (int kt = 0; kt < 2; ++kt)
#pragma unroll
        for (int nt = 0; nt < 2; ++nt) B2f[kt][nt] = load_b_frag(W2, H, HD, nt, kt, lane);
#pragma unroll
    for (int nt = 0; nt < 4; ++nt) B3f[nt] = load_b_frag(W3, 2 * H, H, nt, 0, lane);
#pragma unroll
    for (int kt = 0; kt < 2; ++kt) B4f[kt] = load_b_frag(W4, 1, 2 * H, 0, kt, lane);

    // Biases, one value per lane per N-tile
    float b1v[4], b2v[2], b3v[4], b4v;
#pragma unroll
    for (int nt = 0; nt < 4; ++nt) { int i = nt * 16 + nl; b1v[nt] = (i < HD) ? b1[i] : 0.0f; }
#pragma unroll
    for (int nt = 0; nt < 2; ++nt) b2v[nt] = b2[nt * 16 + nl];
#pragma unroll
    for (int nt = 0; nt < 4; ++nt) b3v[nt] = b3[nt * 16 + nl];
    b4v = (nl == 0) ? b4[0] : 0.0f;

    // Zero the XH pad columns 48..63 (never written again)
    {
        v8h z = {};
        *(v8h*)(XH + row * PITCH + 48 + hsel * 8) = z;
    }

    // h carry in C-fragment layout (2 N-tiles of 16 cols), f32 accumulators
    float hc[2][8];
#pragma unroll
    for (int nt = 0; nt < 2; ++nt)
#pragma unroll
        for (int r = 0; r < 8; ++r) hc[nt][r] = 0.0f;

    for (int t = 0; t < T; ++t) {
        // x tile -> XH cols 0..15, converted to packed f16 (one 16B LDS store per lane)
        {
            const float* xsrc = x + ((size_t)(base + row) * T + t) * D + hsel * 8;
            v4f x0 = *(const v4f*)(xsrc);
            v4f x1 = *(const v4f*)(xsrc + 4);
            v8h xh;
#pragma unroll
            for (int i = 0; i < 4; ++i) {
                xh[i]     = (_Float16)x0[i];
                xh[i + 4] = (_Float16)x1[i];
            }
            *(v8h*)(XH + row * PITCH + hsel * 8) = xh;
        }
        if (lane < 16) {
            size_t bi = (size_t)(base + lane) * T + t;
            float d0 = dt[bi * 2 + 0], d1 = dt[bi * 2 + 1];
            SC[lane] = (d1 - d0) * DT_SCALER;
            Y[lane]  = x[bi * D + 0];     // y init = x[:,0:1]
        }
        if (t + 1 < T)   // warm the next step's x tile into cache
            __builtin_prefetch(x + ((size_t)(base + row) * T + (t + 1)) * D + hsel * 8, 0, 0);
        LDS_FENCE();

        float scv[8];
#pragma unroll
        for (int r = 0; r < 8; ++r) scv[r] = SC[mb + r];

        for (int e = 0; e < EULER; ++e) {
            // spill h (C layout, cvt to f16) -> XH cols 16..47
#pragma unroll
            for (int nt = 0; nt < 2; ++nt)
#pragma unroll
                for (int r = 0; r < 8; ++r)
                    XH[(mb + r) * PITCH + 16 + nt * 16 + nl] = (_Float16)hc[nt][r];
            LDS_FENCE();

            // A-fragments from XH: both chains' inputs up front
            v16h a3  = load_a_frag(XH + row * PITCH + 16 + hsel * 16);  // h window
            v16h a10 = load_a_frag(XH + row * PITCH + 0  + hsel * 16);  // [x|h] low K
            v16h a11 = load_a_frag(XH + row * PITCH + 32 + hsel * 16);  // [x|h] high K

            // ---- layer 3: Z3 = tanh(h @ W3^T + b3), 64 cols ----
#pragma unroll
            for (int nt = 0; nt < 4; ++nt) {
                v8f c = {};
                c = wmma16(a3, B3f[nt], c);
#pragma unroll
                for (int r = 0; r < 8; ++r)
                    Z3[(mb + r) * PITCH + nt * 16 + nl] =
                        (_Float16)vtanh(c[r] + b3v[nt]);
            }

            // ---- layer 1: Z1(cols 0..49) = tanh([x|h] @ W1^T + b1) ----
#pragma unroll
            for (int nt = 0; nt < 4; ++nt) {
                v8f c = {};
                c = wmma16(a10, B1f[0][nt], c);
                c = wmma16(a11, B1f[1][nt], c);
                const bool wr = (nt * 16 + nl) < HD;
#pragma unroll
                for (int r = 0; r < 8; ++r)
                    if (wr) Z1[(mb + r) * PITCH + nt * 16 + nl] =
                                (_Float16)vtanh(c[r] + b1v[nt]);
            }
            LDS_FENCE();

            // ---- layer 4: dy = (Z3 @ W4^T + b4) * scale ; y += STEP*dy ----
            {
                v16h a40 = load_a_frag(Z3 + row * PITCH + 0  + hsel * 16);
                v16h a41 = load_a_frag(Z3 + row * PITCH + 32 + hsel * 16);
                v8f c = {};
                c = wmma16(a40, B4f[0], c);
                c = wmma16(a41, B4f[1], c);
                if (nl == 0) {
#pragma unroll
                    for (int r = 0; r < 8; ++r)
                        Y[mb + r] += STEP * ((c[r] + b4v) * scv[r]);
                }
            }

            // ---- layer 2: dh = tanh(Z1 @ W2^T + b2) * scale ; h += STEP*dh ----
            {
                v16h a20 = load_a_frag(Z1 + row * PITCH + 0  + hsel * 16);
                v16h a21 = load_a_frag(Z1 + row * PITCH + 32 + hsel * 16);
#pragma unroll
                for (int nt = 0; nt < 2; ++nt) {
                    v8f c = {};
                    c = wmma16(a20, B2f[0][nt], c);
                    c = wmma16(a21, B2f[1][nt], c);   // stale Z1 cols>=50 are zeroed by B2f
#pragma unroll
                    for (int r = 0; r < 8; ++r)
                        hc[nt][r] += STEP * (vtanh(c[r] + b2v[nt]) * scv[r]);
                }
            }
        }  // euler

        LDS_FENCE();
        if (lane < 16) out[(size_t)(base + lane) * T + t] = Y[lane];
        LDS_FENCE();
    }  // t
}

extern "C" void kernel_launch(void* const* d_in, const int* in_sizes, int n_in,
                              void* d_out, int out_size, void* d_ws, size_t ws_size,
                              hipStream_t stream) {
    const float* dt = (const float*)d_in[0];
    const float* x  = (const float*)d_in[1];
    const float* W1 = (const float*)d_in[2];
    const float* b1 = (const float*)d_in[3];
    const float* W2 = (const float*)d_in[4];
    const float* b2 = (const float*)d_in[5];
    const float* W3 = (const float*)d_in[6];
    const float* b3 = (const float*)d_in[7];
    const float* W4 = (const float*)d_in[8];
    const float* b4 = (const float*)d_in[9];
    float* out = (float*)d_out;

    dim3 grid(BATCH / (16 * WPB));   // 8192 / 64 = 128 blocks (512 waves, 16 rows each)
    dim3 block(WPB * 32);
    hipLaunchKernelGGL(latent_ode_kernel, grid, block, 0, stream,
                       dt, x, W1, b1, W2, b2, W3, b3, W4, b4, out);
}
